// CompGraphConv_24627342475453
// MI455X (gfx1250) — matrix operand order repaired and checked by reading, not compiled, not run on hardware
//
#include <hip/hip_runtime.h>

typedef __attribute__((ext_vector_type(16))) __bf16 v16bf;
typedef __attribute__((ext_vector_type(8)))  __bf16 v8bf;
typedef __attribute__((ext_vector_type(8)))  float  v8f;

#define IN_DIM 128
#define REL    3
#define KPAD   (IN_DIM * REL)   // 384: zero-padded K, one 128-segment per relation
#define TILE_E 16

// ---------------------------------------------------------------------------
// Kernel 0: convert weights to bf16 in workspace.
// Layout: wrel[r][n][k] = W_r[n*128+k]  (B matrix is read column n, contiguous k)
//         whb[n][k]     = Wh[n*128+k]
// ---------------------------------------------------------------------------
__global__ __launch_bounds__(256)
void cgc_cvt_weights(const float* __restrict__ W0, const float* __restrict__ W1,
                     const float* __restrict__ W2, const float* __restrict__ Wh,
                     __bf16* __restrict__ wrel, __bf16* __restrict__ whb) {
  int i = blockIdx.x * 256 + threadIdx.x;
  if (i < IN_DIM * IN_DIM) {
    wrel[i]                     = (__bf16)W0[i];
    wrel[IN_DIM * IN_DIM + i]   = (__bf16)W1[i];
    wrel[2 * IN_DIM * IN_DIM + i] = (__bf16)W2[i];
    whb[i]                      = (__bf16)Wh[i];
  }
}

// ---------------------------------------------------------------------------
// Kernel 1: out[n,:] = n_feats[n,:] @ Wh^T + bh   (initializes d_out)
// 16 nodes per block; 8 waves, wave w owns output columns [16w, 16w+16).
// ---------------------------------------------------------------------------
__global__ __launch_bounds__(256)
void cgc_node_gemm(const float* __restrict__ nf, const __bf16* __restrict__ whb,
                   const float* __restrict__ bh, float* __restrict__ out,
                   int n_nodes) {
  __shared__ __align__(16) __bf16 a_lds[16][IN_DIM];

  const int tid   = threadIdx.x;
  const int node0 = blockIdx.x * 16;

  { // stage 16 node rows as bf16 in LDS (thread t: row t/16, 8-float chunk t%16)
    int r = tid >> 4, c = tid & 15;
    int node = node0 + r;
    int nn = node < n_nodes ? node : (n_nodes - 1);
    const float4* p = (const float4*)(nf + (size_t)nn * IN_DIM + c * 8);
    float4 f0 = p[0], f1 = p[1];
    v8bf av;
    av[0] = (__bf16)f0.x; av[1] = (__bf16)f0.y; av[2] = (__bf16)f0.z; av[3] = (__bf16)f0.w;
    av[4] = (__bf16)f1.x; av[5] = (__bf16)f1.y; av[6] = (__bf16)f1.z; av[7] = (__bf16)f1.w;
    *(v8bf*)&a_lds[r][c * 8] = av;
  }
  __syncthreads();

  const int lane = tid & 31, wave = tid >> 5;
  const int n0 = wave << 4;
  const int l15 = lane & 15, khalf = lane >> 4;   // A: row=l15; B: col=l15

  v8f acc = {};
#pragma unroll
  for (int kk = 0; kk < IN_DIM; kk += 32) {
    union { v16bf v; uint4 q[2]; } A, B;
    // A frag: 16 contiguous K starting at kk + 16*khalf
    A.q[0] = *(const uint4*)&a_lds[l15][kk + khalf * 16];
    A.q[1] = *(const uint4*)&a_lds[l15][kk + khalf * 16 + 8];
    // B frag: column n0+l15, 16 contiguous K starting at kk + 16*khalf
    const int kg = kk + khalf * 16;
    const uint4* bp = (const uint4*)(whb + (size_t)(n0 + l15) * IN_DIM + kg);
    B.q[0] = bp[0]; B.q[1] = bp[1];
    acc = __builtin_amdgcn_wmma_f32_16x16x32_bf16(false, A.v, false, B.v,
                                                  (short)0, acc, false, false);
  }

  const float bias = bh[n0 + l15];
#pragma unroll
  for (int j = 0; j < 8; ++j) {
    int m = khalf * 8 + j;             // C/D layout: VGPR j -> M=j (+8 for hi half)
    int node = node0 + m;
    if (node < n_nodes)
      out[(size_t)node * IN_DIM + n0 + l15] = acc[j] + bias;
  }
}

// ---------------------------------------------------------------------------
// Kernel 2: fused edge GEMM + scatter.
// 16 edges per block. A tile is 16x384 bf16, zero-padded; row e carries
// (h_src - h_dst) in segment [128*etype, 128*etype+128). B = [W0^T;W1^T;W2^T].
// Epilogue: atomic f32 add of (h + b_etype) into out[dst].
// ---------------------------------------------------------------------------
__global__ __launch_bounds__(256)
void cgc_edge_gemm_scatter(const float* __restrict__ nf,
                           const int* __restrict__ src, const int* __restrict__ dst,
                           const int* __restrict__ eft,
                           const __bf16* __restrict__ wrel,
                           const float* __restrict__ b0, const float* __restrict__ b1,
                           const float* __restrict__ b2,
                           float* __restrict__ out, int n_edges) {
  __shared__ __align__(16) __bf16 a_lds[16][KPAD];
  __shared__ int src_s[16], dst_s[16], et_s[16], ok_s[16];
  __shared__ float bias_s[REL][IN_DIM];

  const int tid = threadIdx.x;
  const int e0  = blockIdx.x * TILE_E;

  // zero-fill padded A tile (3072 dwords across 256 threads)
  unsigned int* az = (unsigned int*)&a_lds[0][0];
#pragma unroll
  for (int i = tid; i < 16 * KPAD / 2; i += 256) az[i] = 0u;

  if (tid < 16) {
    int e  = e0 + tid;
    int ok = e < n_edges;
    int s  = ok ? src[e] : 0;
    int d  = ok ? dst[e] : 0;
    int et = ok ? eft[e] : 0;
    int etv = (et >= 0 && et < REL);
    src_s[tid] = s;
    dst_s[tid] = d;
    et_s[tid]  = etv ? et : 0;
    ok_s[tid]  = ok && etv;
  }
  if (tid < IN_DIM) {
    bias_s[0][tid] = b0[tid];
    bias_s[1][tid] = b1[tid];
    bias_s[2][tid] = b2[tid];
  }
  __syncthreads();

  { // gather diff, convert to bf16, drop into this edge's relation segment
    int r = tid >> 4, c = tid & 15;
    const float4* ps = (const float4*)(nf + (size_t)src_s[r] * IN_DIM + c * 8);
    const float4* pd = (const float4*)(nf + (size_t)dst_s[r] * IN_DIM + c * 8);
    float4 s0 = ps[0], s1 = ps[1];
    float4 d0 = pd[0], d1 = pd[1];
    v8bf av;
    av[0] = (__bf16)(s0.x - d0.x); av[1] = (__bf16)(s0.y - d0.y);
    av[2] = (__bf16)(s0.z - d0.z); av[3] = (__bf16)(s0.w - d0.w);
    av[4] = (__bf16)(s1.x - d1.x); av[5] = (__bf16)(s1.y - d1.y);
    av[6] = (__bf16)(s1.z - d1.z); av[7] = (__bf16)(s1.w - d1.w);
    *(v8bf*)&a_lds[r][et_s[r] * IN_DIM + c * 8] = av;
  }
  __syncthreads();

  const int lane = tid & 31, wave = tid >> 5;
  const int n0 = wave << 4;
  const int l15 = lane & 15, khalf = lane >> 4;

  v8f acc = {};
#pragma unroll
  for (int kk = 0; kk < KPAD; kk += 32) {     // 12 WMMAs over padded K=384
    union { v16bf v; uint4 q[2]; } A, B;
    A.q[0] = *(const uint4*)&a_lds[l15][kk + khalf * 16];
    A.q[1] = *(const uint4*)&a_lds[l15][kk + khalf * 16 + 8];
    const int kg  = kk + khalf * 16;          // global K index (multiple of 16)
    const int rel = kg >> 7;                  // which W this 16-chunk lives in
    const int kin = kg & (IN_DIM - 1);
    const uint4* bp =
        (const uint4*)(wrel + (((size_t)rel * IN_DIM + n0 + l15) << 7) + kin);
    B.q[0] = bp[0]; B.q[1] = bp[1];
    acc = __builtin_amdgcn_wmma_f32_16x16x32_bf16(false, A.v, false, B.v,
                                                  (short)0, acc, false, false);
  }

  const int col = n0 + l15;
#pragma unroll
  for (int j = 0; j < 8; ++j) {
    int m = khalf * 8 + j;                    // edge row within the tile
    if (ok_s[m]) {
      float v = acc[j] + bias_s[et_s[m]][col];
      atomicAdd(out + (size_t)dst_s[m] * IN_DIM + col, v);
    }
  }
}

// ---------------------------------------------------------------------------
extern "C" void kernel_launch(void* const* d_in, const int* in_sizes, int n_in,
                              void* d_out, int out_size, void* d_ws, size_t ws_size,
                              hipStream_t stream) {
  const float* nf  = (const float*)d_in[0];
  const int*   src = (const int*)d_in[1];
  const int*   dst = (const int*)d_in[2];
  const int*   eft = (const int*)d_in[3];
  const float* W0  = (const float*)d_in[4];
  const float* b0  = (const float*)d_in[5];
  const float* W1  = (const float*)d_in[6];
  const float* b1  = (const float*)d_in[7];
  const float* W2  = (const float*)d_in[8];
  const float* b2  = (const float*)d_in[9];
  const float* Wh  = (const float*)d_in[10];
  const float* bh  = (const float*)d_in[11];
  float* out = (float*)d_out;

  const int n_nodes = in_sizes[0] / IN_DIM;
  const int n_edges = in_sizes[1];

  // workspace: 3*128*128 + 128*128 bf16 = 128 KB
  __bf16* wrel = (__bf16*)d_ws;
  __bf16* whb  = wrel + (size_t)REL * IN_DIM * IN_DIM;

  cgc_cvt_weights<<<(IN_DIM * IN_DIM + 255) / 256, 256, 0, stream>>>(
      W0, W1, W2, Wh, wrel, whb);
  cgc_node_gemm<<<(n_nodes + 15) / 16, 256, 0, stream>>>(
      nf, whb, bh, out, n_nodes);
  cgc_edge_gemm_scatter<<<(n_edges + 15) / 16, 256, 0, stream>>>(
      nf, src, dst, eft, wrel, b0, b1, b2, out, n_edges);
}
